// MultiHeadSelfAttention_26199300506139
// MI455X (gfx1250) — matrix operand order repaired and checked by reading, not compiled, not run on hardware
//
#include <hip/hip_runtime.h>
#include <math.h>

// ---------------------------------------------------------------------------
// MI455X (gfx1250) fused MHA, all matmuls on v_wmma_f32_16x16x32_bf16.
// B=4, S=2048, D=1024, H=16, DK=64. Compute-bound (~137 GFLOP vs ~90MB I/O).
// GEMMs: 128x128x32 block tiles, 8 waves (2x4), 64x32 per wave = 8 WMMAs per
// 6 LDS fragment loads; register double-buffering of global->LDS staging.
// Flash attention: 1 wave / 16-query tile, 32-key steps, online softmax,
// V kept transposed so PV B-fragments are contiguous b128 loads; prefetch
// next K/V tiles (L2-resident) via global_prefetch_b8. Softcap uses the
// native V_TANH_F32 trans op (branchless fallback otherwise).
// Workspace: ~92.3 MB.
// ---------------------------------------------------------------------------

typedef __bf16 bf16;
typedef __attribute__((ext_vector_type(16))) __bf16 v16bf;
typedef __attribute__((ext_vector_type(8)))  float  v8f;

#define WMMA_BF16(A, Bm, C) \
  __builtin_amdgcn_wmma_f32_16x16x32_bf16(false, (A), false, (Bm), (short)0, (C), false, false)

union FragU { uint4 u[2]; v16bf v; };

static constexpr int BATCH = 4;
static constexpr int SEQ   = 2048;
static constexpr int DM    = 1024;
static constexpr int NH    = 16;
static constexpr int DK    = 64;

// Branch-free tanh: prefer the gfx1250 V_TANH_F32 trans instruction.
__device__ __forceinline__ float fast_tanh(float x) {
#if __has_builtin(__builtin_amdgcn_tanhf)
  return __builtin_amdgcn_tanhf(x);
#elif __has_builtin(__builtin_amdgcn_tanh_f32)
  return __builtin_amdgcn_tanh_f32(x);
#else
  // tanh(x) = 1 - 2/(exp(2x)+1)  -- single v_exp_f32 + rcp, no divergence.
  float e = __expf(2.0f * x);
  return 1.0f - 2.0f / (e + 1.0f);
#endif
}

// ---- fragment loaders -----------------------------------------------------
// A-matrix 16x32 bf16: lane L holds row M=L&15; K values are 16 consecutive
// starting at (L<16 ? 0 : 16).
__device__ __forceinline__ v16bf load_a_frag(const bf16* base, int stride, int lane) {
  int row = lane & 15;
  int k0  = (lane < 16) ? 0 : 16;
  const bf16* p = base + row * stride;
  FragU f;
  f.u[0] = *reinterpret_cast<const uint4*>(p + k0);
  f.u[1] = *reinterpret_cast<const uint4*>(p + k0 + 8);
  return f.v;
}

// B-matrix 32x16 bf16: lane L holds column N=L&15; K values are 16 consecutive
// starting at (L<16 ? 0 : 16).
__device__ __forceinline__ v16bf load_b_frag(const bf16* base, int stride, int lane) {
  int row = lane & 15;               // N index (B stored as N-major rows of K)
  int k0  = (lane < 16) ? 0 : 16;
  const uint4* p = reinterpret_cast<const uint4*>(base + row * stride + k0);
  FragU f;
  f.u[0] = p[0];
  f.u[1] = p[1];
  return f.v;
}

// ---- f32 -> bf16 convert --------------------------------------------------
__global__ void cvt_f32_bf16(const float* __restrict__ in, bf16* __restrict__ out, int n) {
  int i = blockIdx.x * blockDim.x + threadIdx.x;
  if (i < n) out[i] = (bf16)in[i];
}

// ---- GEMM: C(M x N) = A(M x 1024) * W(N x 1024)^T, bf16 in, f32 acc -------
// MODE 0: QKV projection epilogue -> scatter q,k to [which][b][h][s][dk],
//         v transposed to [b,h][dk][s].
// MODE 1: output projection epilogue -> fp32 row-major out[m*1024+n].
template <int MODE>
__global__ __launch_bounds__(256) void gemm_bf16(const bf16* __restrict__ A,
                                                 const bf16* __restrict__ W,
                                                 bf16* __restrict__ qkb,
                                                 bf16* __restrict__ vT,
                                                 float* __restrict__ out) {
  constexpr int K = DM;
  __shared__ bf16 As[128 * 40];
  __shared__ bf16 Bs[128 * 40];
  const int tid = threadIdx.x, lane = tid & 31, wave = tid >> 5;
  const int wm = wave >> 2, wn = wave & 3;          // 2 x 4 wave grid
  const long mblock = (long)blockIdx.y * 128;
  const int  nblock = blockIdx.x * 128;

  v8f acc[4][2];
#pragma unroll
  for (int a = 0; a < 4; ++a)
#pragma unroll
    for (int b = 0; b < 2; ++b) acc[a][b] = (v8f){0,0,0,0,0,0,0,0};

  const int row0 = tid >> 2, row1 = (tid + 256) >> 2, seg = tid & 3;

  // register double-buffer of the global->LDS staging
  uint4 ra0, ra1, rb0, rb1;
  ra0 = *reinterpret_cast<const uint4*>(A + (mblock + row0) * K + seg * 8);
  ra1 = *reinterpret_cast<const uint4*>(A + (mblock + row1) * K + seg * 8);
  rb0 = *reinterpret_cast<const uint4*>(W + (long)(nblock + row0) * K + seg * 8);
  rb1 = *reinterpret_cast<const uint4*>(W + (long)(nblock + row1) * K + seg * 8);

  for (int kt = 0; kt < K / 32; ++kt) {
    *reinterpret_cast<uint4*>(&As[row0 * 40 + seg * 8]) = ra0;
    *reinterpret_cast<uint4*>(&As[row1 * 40 + seg * 8]) = ra1;
    *reinterpret_cast<uint4*>(&Bs[row0 * 40 + seg * 8]) = rb0;
    *reinterpret_cast<uint4*>(&Bs[row1 * 40 + seg * 8]) = rb1;
    __syncthreads();

    if (kt + 1 < K / 32) {                         // prefetch next K slice
      int ko = (kt + 1) * 32 + seg * 8;
      ra0 = *reinterpret_cast<const uint4*>(A + (mblock + row0) * K + ko);
      ra1 = *reinterpret_cast<const uint4*>(A + (mblock + row1) * K + ko);
      rb0 = *reinterpret_cast<const uint4*>(W + (long)(nblock + row0) * K + ko);
      rb1 = *reinterpret_cast<const uint4*>(W + (long)(nblock + row1) * K + ko);
    }

    v16bf af[4], bfr[2];
#pragma unroll
    for (int mi = 0; mi < 4; ++mi) af[mi] = load_a_frag(&As[(wm * 64 + mi * 16) * 40], 40, lane);
#pragma unroll
    for (int ni = 0; ni < 2; ++ni) bfr[ni] = load_b_frag(&Bs[(wn * 32 + ni * 16) * 40], 40, lane);
#pragma unroll
    for (int mi = 0; mi < 4; ++mi)
#pragma unroll
      for (int ni = 0; ni < 2; ++ni)
        acc[mi][ni] = WMMA_BF16(af[mi], bfr[ni], acc[mi][ni]);
    __syncthreads();
  }

  const int halfoff = (lane < 16) ? 0 : 8;
#pragma unroll
  for (int mi = 0; mi < 4; ++mi)
#pragma unroll
    for (int ni = 0; ni < 2; ++ni) {
      const long mbase = mblock + wm * 64 + mi * 16 + halfoff;
      const int  nsub  = nblock + wn * 32 + ni * 16;        // 16-aligned, uniform
      const int  n     = nsub + (lane & 15);
      if (MODE == 1) {
#pragma unroll
        for (int i = 0; i < 8; ++i) out[(mbase + i) * DM + n] = acc[mi][ni][i];
      } else {
        const int which = nsub >> 10;                        // uniform per subtile
        const int r = n & 1023, h = r >> 6, d = r & 63;
        if (which == 2) {
#pragma unroll
          for (int i = 0; i < 8; ++i) {
            long m = mbase + i;
            int b = (int)(m >> 11), s = (int)(m & 2047);
            vT[(((long)(b * NH + h)) * DK + d) * SEQ + s] = (bf16)acc[mi][ni][i];
          }
        } else {
#pragma unroll
          for (int i = 0; i < 8; ++i) {
            long m = mbase + i;
            int b = (int)(m >> 11), s = (int)(m & 2047);
            qkb[(((long)((which * BATCH + b) * NH + h)) * SEQ + s) * DK + d] =
                (bf16)acc[mi][ni][i];
          }
        }
      }
    }
}

// ---- RoPE on q and k (in place, bf16 storage, f32 math) -------------------
__global__ void rope_kernel(bf16* __restrict__ qk) {
  long idx = (long)blockIdx.x * blockDim.x + threadIdx.x;   // 2*B*H*S*32 pairs
  const long total = 2L * BATCH * NH * SEQ * 32;
  if (idx >= total) return;
  int  d2 = (int)(idx & 31);
  long t  = idx >> 5;                 // row over [2][B][H][S]
  int  s  = (int)(t & (SEQ - 1));
  bf16* p = qk + t * DK + d2 * 2;
  float inv_freq = __expf(-(float)d2 * (9.210340372f / 32.0f));  // 10000^(-d2/32)
  float ang = (float)s * inv_freq;
  float sn, cs;
  __sincosf(ang, &sn, &cs);
  float x1 = (float)p[0], x2 = (float)p[1];
  p[0] = (bf16)(x1 * cs - x2 * sn);
  p[1] = (bf16)(x1 * sn + x2 * cs);
}

// ---- Flash attention: one wave per 16-query tile, 32-key steps ------------
__global__ __launch_bounds__(128) void flash_attn(const bf16* __restrict__ qkb,
                                                  const bf16* __restrict__ vT,
                                                  bf16* __restrict__ attn) {
  __shared__ bf16 Ps[4][16 * 40];                     // per-wave P staging
  const int lane = threadIdx.x & 31;
  const int wid  = blockIdx.x * 4 + (threadIdx.x >> 5);   // 0..8191
  const int qt   = wid & (SEQ / 16 - 1);
  const int bh   = wid >> 7;
  const int qbase = qt * 16;
  const bf16* qptr = qkb + (long)bh * SEQ * DK;
  const bf16* kptr = qkb + (long)(BATCH * NH + bh) * SEQ * DK;
  const bf16* vptr = vT  + (long)bh * DK * SEQ;
  bf16* lds = Ps[threadIdx.x >> 5];

  // Q A-fragments (Dk = 64 -> two 16x32 fragments), loaded once.
  v16bf qa[2];
  {
    const bf16* qr = qptr + (long)(qbase + (lane & 15)) * DK;
    int k0 = (lane < 16) ? 0 : 16;
#pragma unroll
    for (int c = 0; c < 2; ++c) {
      FragU f;
      f.u[0] = *reinterpret_cast<const uint4*>(qr + c * 32 + k0);
      f.u[1] = *reinterpret_cast<const uint4*>(qr + c * 32 + k0 + 8);
      qa[c] = f.v;
    }
  }

  float rowmax[8], rowsum[8];
  v8f o[4];
#pragma unroll
  for (int i = 0; i < 8; ++i) { rowmax[i] = -INFINITY; rowsum[i] = 0.f; }
#pragma unroll
  for (int t = 0; t < 4; ++t) o[t] = (v8f){0,0,0,0,0,0,0,0};

  const int halfoff = (lane < 16) ? 0 : 8;
  const int nkt = (qbase + 16 + 31) >> 5;             // 32-key tiles (causal bound)
  const float scale = 0.125f;                          // 1/sqrt(64)

  for (int kt2 = 0; kt2 < nkt; ++kt2) {
    const int kbase = kt2 * 32;

    if (kt2 + 1 < nkt) {                              // warm next K/V tiles
      __builtin_prefetch(kptr + (long)(kbase + 32 + lane) * DK, 0, 3);
      __builtin_prefetch(vptr + (long)lane * SEQ + kbase + 32, 0, 3);
      __builtin_prefetch(vptr + (long)(lane + 32) * SEQ + kbase + 32, 0, 3);
    }

    // ---- S = Q K^T for 32 keys (two 16-key C tiles, 4 WMMAs) ----
    v8f s0 = (v8f){0,0,0,0,0,0,0,0};
    v8f s1 = (v8f){0,0,0,0,0,0,0,0};
    {
      int k0 = (lane < 16) ? 0 : 16;
#pragma unroll
      for (int n16 = 0; n16 < 2; ++n16) {
        const bf16* kr = kptr + (long)(kbase + n16 * 16 + (lane & 15)) * DK;
#pragma unroll
        for (int kc = 0; kc < 2; ++kc) {
          FragU f;
          const uint4* p = reinterpret_cast<const uint4*>(kr + kc * 32 + k0);
          f.u[0] = p[0];
          f.u[1] = p[1];
          if (n16 == 0) s0 = WMMA_BF16(qa[kc], f.v, s0);
          else          s1 = WMMA_BF16(qa[kc], f.v, s1);
        }
      }
    }

    // ---- softcap + causal mask + online softmax (rows in 16-lane halves) --
    const int key0 = kbase + (lane & 15);
#pragma unroll
    for (int i = 0; i < 8; ++i) {
      int qrow = qbase + i + halfoff;
      float a = 50.f * fast_tanh(s0[i] * scale * 0.02f);
      if (key0 > qrow) a = -1e30f;
      float b = 50.f * fast_tanh(s1[i] * scale * 0.02f);
      if (key0 + 16 > qrow) b = -1e30f;

      float mx = fmaxf(a, b);
#pragma unroll
      for (int d = 8; d >= 1; d >>= 1) mx = fmaxf(mx, __shfl_xor(mx, d, 16));
      float mN = fmaxf(rowmax[i], mx);
      float alpha = __expf(rowmax[i] - mN);
      rowmax[i] = mN;
      float p0 = __expf(a - mN);
      float p1 = __expf(b - mN);
      float rs = p0 + p1;
#pragma unroll
      for (int d = 8; d >= 1; d >>= 1) rs += __shfl_xor(rs, d, 16);
      rowsum[i] = rowsum[i] * alpha + rs;
#pragma unroll
      for (int t = 0; t < 4; ++t) o[t][i] *= alpha;

      lds[(i + halfoff) * 40 + (lane & 15)]      = (bf16)p0;
      lds[(i + halfoff) * 40 + 16 + (lane & 15)] = (bf16)p1;
    }
    asm volatile("s_wait_dscnt 0x0" ::: "memory");    // wave-local LDS RAW

    // ---- O += P V  (P 16x32 via LDS A-layout; V^T -> contiguous B frags) --
    v16bf pa;
    {
      int row = lane & 15;
      int k0  = (lane < 16) ? 0 : 16;
      FragU f;
      f.u[0] = *reinterpret_cast<const uint4*>(lds + row * 40 + k0);
      f.u[1] = *reinterpret_cast<const uint4*>(lds + row * 40 + k0 + 8);
      pa = f.v;
    }
    {
      int k0 = (lane < 16) ? 0 : 16;
#pragma unroll
      for (int t = 0; t < 4; ++t) {
        const bf16* vr = vptr + (long)(t * 16 + (lane & 15)) * SEQ + kbase + k0;
        FragU f;
        const uint4* p = reinterpret_cast<const uint4*>(vr);
        f.u[0] = p[0];
        f.u[1] = p[1];
        o[t] = WMMA_BF16(pa, f.v, o[t]);
      }
    }
  }

  // ---- finalize: O /= rowsum, scatter to [b][s][h*64+d] bf16 --------------
  const int b = bh >> 4, h = bh & 15;
#pragma unroll
  for (int t = 0; t < 4; ++t)
#pragma unroll
    for (int i = 0; i < 8; ++i) {
      int qrow = qbase + i + halfoff;
      float val = o[t][i] / rowsum[i];
      attn[((long)(b * SEQ + qrow)) * DM + h * DK + t * 16 + (lane & 15)] = (bf16)val;
    }
}

// ---------------------------------------------------------------------------
extern "C" void kernel_launch(void* const* d_in, const int* in_sizes, int n_in,
                              void* d_out, int out_size, void* d_ws, size_t ws_size,
                              hipStream_t stream) {
  (void)in_sizes; (void)n_in; (void)out_size; (void)ws_size;
  const float* x     = (const float*)d_in[0];   // (4,2048,1024)
  const float* w_qkv = (const float*)d_in[1];   // (3072,1024)
  const float* w_out = (const float*)d_in[2];   // (1024,1024)
  float* out = (float*)d_out;

  char* ws = (char*)d_ws;
  bf16* xb    = (bf16*)ws;  ws += (size_t)BATCH * SEQ * DM * 2;            // 16 MB
  bf16* wqkvb = (bf16*)ws;  ws += (size_t)3 * DM * DM * 2;                 // 6 MB
  bf16* woutb = (bf16*)ws;  ws += (size_t)DM * DM * 2;                     // 2 MB
  bf16* qkb   = (bf16*)ws;  ws += (size_t)2 * BATCH * NH * SEQ * DK * 2;   // 32 MB (q,k)
  bf16* vT    = (bf16*)ws;  ws += (size_t)BATCH * NH * DK * SEQ * 2;       // 16 MB
  bf16* attn  = (bf16*)ws;  ws += (size_t)BATCH * SEQ * DM * 2;            // 16 MB

  const int nx = BATCH * SEQ * DM;
  const int nq = 3 * DM * DM;
  const int no = DM * DM;
  cvt_f32_bf16<<<(nx + 255) / 256, 256, 0, stream>>>(x, xb, nx);
  cvt_f32_bf16<<<(nq + 255) / 256, 256, 0, stream>>>(w_qkv, wqkvb, nq);
  cvt_f32_bf16<<<(no + 255) / 256, 256, 0, stream>>>(w_out, woutb, no);

  gemm_bf16<0><<<dim3(3 * DM / 128, BATCH * SEQ / 128), 256, 0, stream>>>(
      xb, wqkvb, qkb, vT, nullptr);

  const long npairs = 2L * BATCH * NH * SEQ * 32;
  rope_kernel<<<(int)((npairs + 255) / 256), 256, 0, stream>>>(qkb);

  flash_attn<<<BATCH * NH * (SEQ / 16) / 4, 128, 0, stream>>>(qkb, vT, attn);

  gemm_bf16<1><<<dim3(DM / 128, BATCH * SEQ / 128), 256, 0, stream>>>(
      attn, woutb, nullptr, nullptr, out);
}
